// AttnCell_69879117906744
// MI455X (gfx1250) — matrix operand order repaired and checked by reading, not compiled, not run on hardware
//
#include <hip/hip_runtime.h>
#include <hip/hip_bf16.h>
#include <math.h>

typedef __attribute__((ext_vector_type(16))) _Float16 v16h;
typedef __attribute__((ext_vector_type(8)))  float    v8f;
typedef unsigned int v4u __attribute__((ext_vector_type(4)));
typedef unsigned int v8u __attribute__((ext_vector_type(8)));

#define WMMA_F16(A, B, C) \
  __builtin_amdgcn_wmma_f32_16x16x32_f16(false, (A), false, (B), (short)0, (C), false, false)

// ---------------------------------------------------------------------------
// CDNA5 async data movement helpers
// ---------------------------------------------------------------------------
__device__ __forceinline__ unsigned lds_off(const void* p) {
  // addrspace(3) flat addresses: low 32 bits are the LDS byte address
  return (unsigned)(unsigned long long)(uintptr_t)p;
}

// 64B per lane via 4 x GLOBAL_LOAD_ASYNC_TO_LDS_B128 (tracked by ASYNCcnt)
__device__ __forceinline__ void async_copy64(unsigned lds_addr, const void* gptr) {
  unsigned long long ga = (unsigned long long)(uintptr_t)gptr;
  asm volatile(
      "global_load_async_to_lds_b128 %0, %1, off\n\t"
      "global_load_async_to_lds_b128 %0, %1, off offset:16\n\t"
      "global_load_async_to_lds_b128 %0, %1, off offset:32\n\t"
      "global_load_async_to_lds_b128 %0, %1, off offset:48"
      :: "v"(lds_addr), "v"(ga) : "memory");
}
__device__ __forceinline__ void wait_async0() {
  asm volatile("s_wait_asynccnt 0x0" ::: "memory");
}
__device__ __forceinline__ void wait_tensor0() {
#if __has_builtin(__builtin_amdgcn_s_wait_tensorcnt)
  __builtin_amdgcn_s_wait_tensorcnt(0);
#else
  asm volatile("s_wait_tensorcnt 0x0" ::: "memory");
#endif
}

// TDM: load a 2D tile (tile_d0 x tile_d1 elements, 2-byte elements) from a
// row-major tensor (row length tensor_d0, row stride stride0 elements) into
// LDS packed as [tile_d1][tile_d0]. One instruction per wave; TENSORcnt.
__device__ __forceinline__ void tdm_load_2d(unsigned lds_addr, const void* gaddr,
                                            unsigned tensor_d0, unsigned tensor_d1,
                                            unsigned tile_d0, unsigned tile_d1,
                                            unsigned long long stride0) {
  unsigned long long ga = (unsigned long long)(uintptr_t)gaddr;
  v4u g0;  // D# group 0
  g0[0] = 1u;                                            // count=1 (valid), user
  g0[1] = lds_addr;                                      // lds_addr
  g0[2] = (unsigned)ga;                                  // global_addr[31:0]
  g0[3] = (unsigned)((ga >> 32) & 0x01ffffffu) | (2u << 30); // addr[56:32], type=2
  v8u g1;  // D# group 1
  g1[0] = 1u << 16;                                      // data_size=1 (2 bytes)
  g1[1] = (tensor_d0 & 0xffffu) << 16;                   // tensor_dim0[15:0]
  g1[2] = (tensor_d0 >> 16) | ((tensor_d1 & 0xffffu) << 16); // d0[31:16] | d1[15:0]
  g1[3] = (tensor_d1 >> 16) | ((tile_d0 & 0xffffu) << 16);   // d1[31:16] | tile_d0
  g1[4] = (tile_d1 & 0xffffu);                           // tile_dim1 (tile_dim2=0)
  g1[5] = (unsigned)(stride0 & 0xffffffffu);             // dim0_stride[31:0]
  g1[6] = (unsigned)((stride0 >> 32) & 0xffffu);         // dim0_stride[47:32]
  g1[7] = 0u;                                            // dim1_stride unused (2D)
  asm volatile("tensor_load_to_lds %0, %1" :: "s"(g0), "s"(g1) : "memory");
}

// ---------------------------------------------------------------------------
// Kernel 0: convert a 256x256 f32 weight matrix to f16 (one-time, 256 KB read)
// ---------------------------------------------------------------------------
__global__ __launch_bounds__(256) void convert_w(const float* __restrict__ W,
                                                 _Float16* __restrict__ Out) {
  const int i = blockIdx.x * 256 + threadIdx.x;   // grid 256 -> 65536 elements
  Out[i] = (_Float16)W[i];
}

// ---------------------------------------------------------------------------
// Kernel 1: projection GEMM.  Out(f16) = (X(f32) @ Wh(f16) + bias) * outScale
//   X: (16384, 256) row-major, Wh: (256, 256) f16 row-major.
//   transpose==0: Out row-major (16384,256);  ==1: (B=4, 256, 4096) K-transposed
// 256 threads = 8 waves; wave computes a 16x256 tile. W K-slabs double-buffered
// in LDS via GLOBAL_LOAD_ASYNC_TO_LDS (ASYNCcnt).
// ---------------------------------------------------------------------------
__global__ __launch_bounds__(256) void qkv_gemm(
    const float* __restrict__ X,
    const _Float16* __restrict__ Wh,
    const float* __restrict__ bias,
    _Float16* __restrict__ Out,
    int transpose, float outScale)
{
  __shared__ __align__(32) _Float16 sW[2][32 * 256];   // 2 x 16 KB W slabs

  const int t    = threadIdx.x;
  const int lane = t & 31;
  const int wave = t >> 5;
  const int half = lane >> 4;
  const int rr   = lane & 15;
  const int rowBase = blockIdx.x * 128 + wave * 16;
  const int wr = t >> 3, wc = (t & 7) * 32;            // this thread's 64B slice

  const v8f vzero = {};
  v8f acc[16];
#pragma unroll
  for (int nt = 0; nt < 16; ++nt) acc[nt] = vzero;

  // prologue: async-stage slab kk=0 into buffer 0
  async_copy64(lds_off(&sW[0][wr * 256 + wc]), Wh + (size_t)wr * 256 + wc);

  for (int kk = 0; kk < 8; ++kk) {
    const int cur = kk & 1;
    wait_async0();           // this thread's slab-copy portion has landed
    __syncthreads();         // everyone's portion visible; prev reads done
    if (kk + 1 < 8)          // overlap next slab copy with compute
      async_copy64(lds_off(&sW[1 - cur][wr * 256 + wc]),
                   Wh + (size_t)((kk + 1) * 32 + wr) * 256 + wc);

    // A fragment (16x32 f16) from f32 X
    v16h a;
    {
      const float* src = X + (size_t)(rowBase + rr) * 256 + kk * 32;
#pragma unroll
      for (int j = 0; j < 8; ++j) {
        a[j]     = (_Float16)src[half * 8 + j];
        a[8 + j] = (_Float16)src[16 + half * 8 + j];
      }
    }
#pragma unroll
    for (int nt = 0; nt < 16; ++nt) {
      v16h b;
#pragma unroll
      for (int j = 0; j < 8; ++j) {
        b[j]     = sW[cur][(half * 8 + j) * 256 + nt * 16 + rr];
        b[8 + j] = sW[cur][(16 + half * 8 + j) * 256 + nt * 16 + rr];
      }
      acc[nt] = WMMA_F16(a, b, acc[nt]);
    }
  }

#pragma unroll
  for (int nt = 0; nt < 16; ++nt) {
#pragma unroll
    for (int v = 0; v < 8; ++v) {
      const int r = rowBase + v + 8 * half;
      const int c = nt * 16 + rr;
      const float val = (acc[nt][v] + bias[c]) * outScale;
      if (!transpose) {
        Out[(size_t)r * 256 + c] = (_Float16)val;
      } else {
        const int bb = r >> 12, n = r & 4095;
        Out[((size_t)bb * 256 + c) * 4096 + n] = (_Float16)val;
      }
    }
  }
}

// ---------------------------------------------------------------------------
// Kernel 2: flash attention.  Q (pre-scaled 1/16), Kt (B,256,4096), V f16.
// mask (B,4096,4096) int32 0/1 read as low byte; mask<=0 -> score := 1e-9.
// 256 threads (8 waves) per block = 128 Q rows of one batch.
// K/V chunks double-buffered in LDS via TDM (tensor_load_to_lds, TENSORcnt):
// wave0 issues the Kt tile, wave1 the V tile, overlapped with compute.
// ---------------------------------------------------------------------------
__global__ __launch_bounds__(256) void flash_attn(
    const _Float16* __restrict__ Q,
    const _Float16* __restrict__ Kt,
    const _Float16* __restrict__ V,
    const int* __restrict__ mask,
    float* __restrict__ Out)
{
  __shared__ __align__(32) _Float16 sKt[2][256 * 32];   // 2 x 16 KB
  __shared__ __align__(32) _Float16 sV[2][32 * 256];    // 2 x 16 KB
  __shared__ __align__(32) _Float16 sP[8 * 16 * 32];    // 8 KB per-wave P tiles

  const int t    = threadIdx.x;
  const int lane = t & 31;
  const int wave = t >> 5;
  const int half = lane >> 4;
  const int rr   = lane & 15;
  const int blk  = blockIdx.x;                    // 128 blocks
  const int b    = blk >> 5;                      // batch 0..3
  const int row0 = (blk & 31) * 128 + wave * 16;  // n1 base for this wave

  const _Float16* KtB = Kt + (size_t)b * 256 * 4096;
  const _Float16* VB  = V  + (size_t)b * 4096 * 256;

  // preload Q fragments for all 8 K-steps
  v16h qa[8];
  {
    const _Float16* src = Q + ((size_t)b * 4096 + row0 + rr) * 256;
#pragma unroll
    for (int kk = 0; kk < 8; ++kk) {
#pragma unroll
      for (int j = 0; j < 8; ++j) {
        qa[kk][j]     = src[kk * 32 + half * 8 + j];
        qa[kk][8 + j] = src[kk * 32 + 16 + half * 8 + j];
      }
    }
  }

  const v8f vzero = {};
  v8f o[16];
#pragma unroll
  for (int nt = 0; nt < 16; ++nt) o[nt] = vzero;
  float m[8], l[8];
#pragma unroll
  for (int v = 0; v < 8; ++v) { m[v] = -INFINITY; l[v] = 0.f; }

  // mask as bytes: values are 0/1 int32, low byte is sufficient (4x less HBM)
  const unsigned char* mrow =
      (const unsigned char*)mask + (((size_t)b * 4096 + row0) * 4096) * 4;

  // prologue: TDM-stage chunk 0 into buffer 0
  if (wave == 0)       tdm_load_2d(lds_off(&sKt[0][0]), KtB,      4096, 256, 32, 256, 4096ull);
  else if (wave == 1)  tdm_load_2d(lds_off(&sV[0][0]),  VB,        256, 4096, 256, 32,  256ull);

  for (int c = 0; c < 128; ++c) {
    const int n2c = c * 32;
    const int cur = c & 1;
    if (wave < 2) wait_tensor0();   // issuing waves wait for their TDM tile
    __syncthreads();                // tile visible to all; prev-buffer reads done
    if (c + 1 < 128) {              // overlap next chunk's TDM with compute
      if (wave == 0)
        tdm_load_2d(lds_off(&sKt[1 - cur][0]), KtB + (n2c + 32),
                    4096, 256, 32, 256, 4096ull);
      else if (wave == 1)
        tdm_load_2d(lds_off(&sV[1 - cur][0]), VB + (size_t)(n2c + 32) * 256,
                    256, 4096, 256, 32, 256ull);
    }

    // S = Q x Kt-chunk : two 16x16 f32 tiles (cols n2c..+15, n2c+16..+31)
    v8f s0 = vzero, s1 = vzero;
#pragma unroll
    for (int kk = 0; kk < 8; ++kk) {
      const _Float16* kb = &sKt[cur][(kk * 32) * 32];
      v16h b0, b1;
#pragma unroll
      for (int j = 0; j < 8; ++j) {
        b0[j]     = kb[(half * 8 + j) * 32 + rr];
        b0[8 + j] = kb[(16 + half * 8 + j) * 32 + rr];
        b1[j]     = kb[(half * 8 + j) * 32 + 16 + rr];
        b1[8 + j] = kb[(16 + half * 8 + j) * 32 + 16 + rr];
      }
      s0 = WMMA_F16(qa[kk], b0, s0);
      s1 = WMMA_F16(qa[kk], b1, s1);
    }

    // mask-fill (1e-9, NOT -inf) + online softmax across the 32 columns
    float p0[8], p1[8], alpha[8];
#pragma unroll
    for (int v = 0; v < 8; ++v) {
      const unsigned char* mp =
          mrow + ((size_t)(v + 8 * half) * 4096 + n2c + rr) * 4;
      const float x0 = (mp[0]  != 0) ? s0[v] : 1e-9f;
      const float x1 = (mp[64] != 0) ? s1[v] : 1e-9f;
      float tm = fmaxf(x0, x1);
      tm = fmaxf(tm, __shfl_xor(tm, 1));
      tm = fmaxf(tm, __shfl_xor(tm, 2));
      tm = fmaxf(tm, __shfl_xor(tm, 4));
      tm = fmaxf(tm, __shfl_xor(tm, 8));
      const float mn = fmaxf(m[v], tm);
      alpha[v] = __expf(m[v] - mn);
      m[v] = mn;
      p0[v] = __expf(x0 - mn);
      p1[v] = __expf(x1 - mn);
      float ps = p0[v] + p1[v];
      ps += __shfl_xor(ps, 1);
      ps += __shfl_xor(ps, 2);
      ps += __shfl_xor(ps, 4);
      ps += __shfl_xor(ps, 8);
      l[v] = l[v] * alpha[v] + ps;
    }

    // rescale accumulator rows
#pragma unroll
    for (int nt = 0; nt < 16; ++nt) {
#pragma unroll
      for (int v = 0; v < 8; ++v) o[nt][v] *= alpha[v];
    }

    // C-layout -> A-layout conversion of P via per-wave LDS bounce
    {
      _Float16* pw = sP + wave * (16 * 32);
#pragma unroll
      for (int v = 0; v < 8; ++v) {
        pw[(v + 8 * half) * 32 + rr]      = (_Float16)p0[v];
        pw[(v + 8 * half) * 32 + 16 + rr] = (_Float16)p1[v];
      }
    }
    v16h pa;
    {
      const _Float16* pr = sP + wave * (16 * 32) + rr * 32;
#pragma unroll
      for (int j = 0; j < 8; ++j) {
        pa[j]     = pr[half * 8 + j];
        pa[8 + j] = pr[16 + half * 8 + j];
      }
    }

    // O += P x V-chunk  (K = 32)
#pragma unroll
    for (int nt = 0; nt < 16; ++nt) {
      v16h vb;
#pragma unroll
      for (int j = 0; j < 8; ++j) {
        vb[j]     = sV[cur][(half * 8 + j) * 256 + nt * 16 + rr];
        vb[8 + j] = sV[cur][(16 + half * 8 + j) * 256 + nt * 16 + rr];
      }
      o[nt] = WMMA_F16(pa, vb, o[nt]);
    }
  }

  // normalize and store f32 output
#pragma unroll
  for (int nt = 0; nt < 16; ++nt) {
#pragma unroll
    for (int v = 0; v < 8; ++v) {
      const int r = row0 + v + 8 * half;
      Out[((size_t)b * 4096 + r) * 256 + nt * 16 + rr] = o[nt][v] / l[v];
    }
  }
}

// ---------------------------------------------------------------------------
extern "C" void kernel_launch(void* const* d_in, const int* in_sizes, int n_in,
                              void* d_out, int out_size, void* d_ws, size_t ws_size,
                              hipStream_t stream) {
  const float* inputs1 = (const float*)d_in[0];   // (4,4096,256) f32
  const float* inputs2 = (const float*)d_in[1];   // (4,4096,256) f32
  const int*   mask    = (const int*)d_in[2];     // (4,4096,4096) i32
  const float* Wq      = (const float*)d_in[3];
  const float* bq      = (const float*)d_in[4];
  const float* Wk      = (const float*)d_in[5];
  const float* bk      = (const float*)d_in[6];
  const float* Wv      = (const float*)d_in[7];
  const float* bv      = (const float*)d_in[8];
  float* out = (float*)d_out;

  // workspace: q/kT/v f16 (8 MB each) + three f16 weight copies (128 KB each)
  _Float16* qf   = (_Float16*)d_ws;
  _Float16* ktf  = qf   + (size_t)16384 * 256;
  _Float16* vf   = ktf  + (size_t)16384 * 256;
  _Float16* wq16 = vf   + (size_t)16384 * 256;
  _Float16* wk16 = wq16 + (size_t)256 * 256;
  _Float16* wv16 = wk16 + (size_t)256 * 256;

  convert_w<<<256, 256, 0, stream>>>(Wq, wq16);
  convert_w<<<256, 256, 0, stream>>>(Wk, wk16);
  convert_w<<<256, 256, 0, stream>>>(Wv, wv16);

  // Q pre-scaled by 1/sqrt(HIDDEN)=1/16; K stored transposed (B,256,4096)
  qkv_gemm<<<128, 256, 0, stream>>>(inputs1, wq16, bq, qf,  0, 0.0625f);
  qkv_gemm<<<128, 256, 0, stream>>>(inputs2, wk16, bk, ktf, 1, 1.0f);
  qkv_gemm<<<128, 256, 0, stream>>>(inputs2, wv16, bv, vf,  0, 1.0f);

  flash_attn<<<128, 256, 0, stream>>>(qf, ktf, vf, mask, out);
}
